// MacroEncoder_62586263437423
// MI455X (gfx1250) — compile-verified, matrix operand order checked by reading, not run on hardware
//
#include <hip/hip_runtime.h>
#include <math.h>

// ---------------- constants ----------------
#define DMODEL   256
#define DSTATE   64
#define DCONV    4
#define DINNER   512
#define NHEADS   8
#define HEADDIM  64
#define DXBC     640          // DINNER + 2*DSTATE
#define DPROJ    1160         // 2*DINNER + 2*DSTATE + NHEADS
#define NLAYERS  4
#define BB       4
#define LL       4096
#define TOK      (BB*LL)      // 16384

typedef __attribute__((ext_vector_type(16))) __bf16 bf16x16;
typedef __attribute__((ext_vector_type(8)))  __bf16 bf16x8;
typedef __attribute__((ext_vector_type(8)))  float  floatx8;

// ---------------- helpers ----------------
__global__ void cvt_bf16_kernel(const float* __restrict__ in, __bf16* __restrict__ out, int n) {
    int i = blockIdx.x * blockDim.x + threadIdx.x;
    if (i < n) out[i] = (__bf16)in[i];
}

// x0 = feat @ projW.T + projb + embed_lat[lat] + embed_lon[lon]
__global__ void embed_kernel(const float* __restrict__ feat, const int* __restrict__ lat,
                             const int* __restrict__ lon, const float* __restrict__ elat,
                             const float* __restrict__ elon, const float* __restrict__ pw,
                             const float* __restrict__ pb, float* __restrict__ x) {
    int idx = blockIdx.x * blockDim.x + threadIdx.x;      // TOK*DMODEL
    int tok = idx >> 8, d = idx & 255;
    float s = pb[d];
    #pragma unroll
    for (int j = 0; j < 10; ++j) s += feat[tok * 10 + j] * pw[d * 10 + j];
    s += elat[lat[tok] * DMODEL + d] + elon[lon[tok] * DMODEL + d];
    x[idx] = s;
}

// ---------------- WMMA GEMM: C[M,N] = A[M,K] @ W[N,K]^T (+C if acc) ----------------
// bf16 inputs, f32 accum.  Each wave owns a 16x64 strip (4 accumulators, A reused
// 4x per K-step).  8 waves/block -> 128x64 block tile.
// All 5 fragment loads of a K-step are issued as one batch BEFORE the 4-wmma
// sequence so the scheduler can use progressive s_wait_loadcnt values and
// overlap later loads with earlier wmmas.
// Boundary note: WMMA D column n depends only on B column n (same lane group),
// so out-of-range columns need no zeroing -- clamp the weight-row pointer to
// valid memory and guard only the stores.
__global__ __launch_bounds__(256) void wmma_gemm_bf16(
    const __bf16* __restrict__ A, const __bf16* __restrict__ W,
    float* __restrict__ C, int M, int N, int K, int acc)
{
    const int lane  = threadIdx.x & 31;
    const int wave  = threadIdx.x >> 5;            // 0..7 -> M sub-tile
    const int tileM = blockIdx.x * 128 + wave * 16;
    const int tileN = blockIdx.y * 64;
    const int row   = tileM + (lane & 15);
    const int khalf = (lane >> 4) << 3;            // K half select: 0 or 8
    const int rsel  = (lane >> 4) << 3;            // C/D row offset for lane group

    floatx8 cfrag[4];
    int coln[4];
    const __bf16* wrow[4];
    #pragma unroll
    for (int t = 0; t < 4; ++t) {
        coln[t] = tileN + t * 16 + (lane & 15);
        wrow[t] = W + (size_t)((coln[t] < N) ? coln[t] : 0) * K;  // clamp, no masking
        #pragma unroll
        for (int j = 0; j < 8; ++j) cfrag[t][j] = 0.f;
        if (acc && coln[t] < N) {
            #pragma unroll
            for (int j = 0; j < 8; ++j)
                cfrag[t][j] = C[(size_t)(tileM + j + rsel) * N + coln[t]];
        }
    }

    const __bf16* arow = A + (size_t)row * K;
    for (int k0 = 0; k0 < K; k0 += 32) {
        // ---- issue all loads for this K-step first ----
        bf16x8 alo = *(const bf16x8*)(arow + k0 + khalf);
        bf16x8 ahi = *(const bf16x8*)(arow + k0 + 16 + khalf);
        bf16x8 blo[4], bhi[4];
        #pragma unroll
        for (int t = 0; t < 4; ++t) {
            blo[t] = *(const bf16x8*)(wrow[t] + k0 + khalf);
            bhi[t] = *(const bf16x8*)(wrow[t] + k0 + 16 + khalf);
        }
        // ---- then run the 4-wmma sequence reusing one A fragment ----
        bf16x16 afrag;
        #pragma unroll
        for (int e = 0; e < 8; ++e) { afrag[e] = alo[e]; afrag[8 + e] = ahi[e]; }
        #pragma unroll
        for (int t = 0; t < 4; ++t) {
            bf16x16 bfrag;
            #pragma unroll
            for (int e = 0; e < 8; ++e) { bfrag[e] = blo[t][e]; bfrag[8 + e] = bhi[t][e]; }
            cfrag[t] = __builtin_amdgcn_wmma_f32_16x16x32_bf16(
                false, afrag, false, bfrag, (short)0, cfrag[t], false, false);
        }
    }

    #pragma unroll
    for (int t = 0; t < 4; ++t) {
        if (coln[t] < N) {
            #pragma unroll
            for (int j = 0; j < 8; ++j)
                C[(size_t)(tileM + j + rsel) * N + coln[t]] = cfrag[t][j];
        }
    }
}

// ---------------- depthwise causal conv (4 tap) + bias + SiLU ----------------
// rev=0: out[l] = sum_j w[j]*x[l-3+j]   rev=1: out[l] = sum_j w[j]*x[l+3-j]
__global__ void conv_silu_kernel(const float* __restrict__ zx, const float* __restrict__ cw,
                                 const float* __restrict__ cb, float* __restrict__ xconv, int rev) {
    int idx = blockIdx.x * blockDim.x + threadIdx.x;
    if (idx >= TOK * DXBC) return;
    int tok = idx / DXBC, c = idx % DXBC;
    int b = tok >> 12, l = tok & (LL - 1);
    const float* w = cw + c * DCONV;
    float s = 0.f;
    if (!rev) {
        #pragma unroll
        for (int j = 0; j < 4; ++j) {
            int ls = l - 3 + j;
            if (ls >= 0) s += w[j] * zx[(size_t)(b * LL + ls) * DPROJ + DINNER + c];
        }
    } else {
        #pragma unroll
        for (int j = 0; j < 4; ++j) {
            int ls = l + 3 - j;
            if (ls < LL) s += w[j] * zx[(size_t)(b * LL + ls) * DPROJ + DINNER + c];
        }
    }
    s += cb[c];
    xconv[idx] = s / (1.f + __expf(-s));
}

// ---------------- dt = softplus(raw + bias); dA = exp(-exp(A_log)*dt) ----------------
__global__ void dt_kernel(const float* __restrict__ zx, const float* __restrict__ dtb,
                          const float* __restrict__ alog, float* __restrict__ dt,
                          float* __restrict__ dA) {
    int idx = blockIdx.x * blockDim.x + threadIdx.x;     // TOK*NHEADS
    int tok = idx >> 3, h = idx & 7;
    float x = zx[(size_t)tok * DPROJ + (DPROJ - NHEADS) + h] + dtb[h];
    float sp = (x > 20.f) ? x : log1pf(__expf(x));
    dt[idx] = sp;
    dA[idx] = __expf(-__expf(alog[h]) * sp);
}

// ---------------- selective scan: one workgroup per (b, head) ----------------
// state h[64p x 64n] held in regs (16 per thread), B/C staged in LDS.
__global__ __launch_bounds__(256) void scan_kernel(
    const float* __restrict__ xconv, const float* __restrict__ dt,
    const float* __restrict__ dA, const float* __restrict__ Dp,
    float* __restrict__ y, int rev)
{
    int b = blockIdx.x >> 3, h = blockIdx.x & 7;
    int tid = threadIdx.x;
    int p = tid >> 2, nb = (tid & 3) << 4;
    __shared__ float sB[DSTATE], sC[DSTATE];
    float hreg[16];
    #pragma unroll
    for (int i = 0; i < 16; ++i) hreg[i] = 0.f;
    float Dv = Dp[h];

    for (int s = 0; s < LL; ++s) {
        int l = rev ? (LL - 1 - s) : s;
        size_t tok = (size_t)b * LL + l;
        if (tid < 64)       sB[tid]      = xconv[tok * DXBC + DINNER + tid];
        else if (tid < 128) sC[tid - 64] = xconv[tok * DXBC + DINNER + DSTATE + (tid - 64)];
        __syncthreads();
        float dtv = dt[tok * NHEADS + h];
        float dAv = dA[tok * NHEADS + h];
        float xp  = xconv[tok * DXBC + h * HEADDIM + p];
        float coef = dtv * xp;
        float part = 0.f;
        #pragma unroll
        for (int i = 0; i < 16; ++i) {
            hreg[i] = hreg[i] * dAv + coef * sB[nb + i];
            part += hreg[i] * sC[nb + i];
        }
        part += __shfl_xor(part, 1, 32);
        part += __shfl_xor(part, 2, 32);
        if ((tid & 3) == 0)
            y[tok * DINNER + h * HEADDIM + p] = part + Dv * xp;
        __syncthreads();
    }
}

// ---------------- y = y * silu(z); RMSNorm; emit bf16 ----------------
__global__ __launch_bounds__(256) void gate_norm_kernel(
    const float* __restrict__ y, const float* __restrict__ zx,
    const float* __restrict__ nw, __bf16* __restrict__ out)
{
    int tok = blockIdx.x, tid = threadIdx.x;
    __shared__ float red[256];
    float v[2], ss = 0.f;
    #pragma unroll
    for (int r = 0; r < 2; ++r) {
        int c = tid + r * 256;
        float z = zx[(size_t)tok * DPROJ + c];
        float val = y[(size_t)tok * DINNER + c] * (z / (1.f + __expf(-z)));
        v[r] = val; ss += val * val;
    }
    red[tid] = ss; __syncthreads();
    for (int o = 128; o > 0; o >>= 1) {
        if (tid < o) red[tid] += red[tid + o];
        __syncthreads();
    }
    float scale = rsqrtf(red[0] / (float)DINNER + 1e-5f);
    #pragma unroll
    for (int r = 0; r < 2; ++r) {
        int c = tid + r * 256;
        out[(size_t)tok * DINNER + c] = (__bf16)(v[r] * scale * nw[c]);
    }
}

// ---------------- mean over L ----------------
__global__ __launch_bounds__(256) void mean_kernel(const float* __restrict__ x, float* __restrict__ out) {
    int b = blockIdx.x >> 8, d = blockIdx.x & 255;
    __shared__ float red[256];
    float s = 0.f;
    for (int l = threadIdx.x; l < LL; l += 256)
        s += x[((size_t)b * LL + l) * DMODEL + d];
    red[threadIdx.x] = s; __syncthreads();
    for (int o = 128; o > 0; o >>= 1) {
        if (threadIdx.x < o) red[threadIdx.x] += red[threadIdx.x + o];
        __syncthreads();
    }
    if (threadIdx.x == 0) out[blockIdx.x] = red[0] / (float)LL;
}

// ---------------- host orchestration ----------------
extern "C" void kernel_launch(void* const* d_in, const int* in_sizes, int n_in,
                              void* d_out, int out_size, void* d_ws, size_t ws_size,
                              hipStream_t stream) {
    const float* feat = (const float*)d_in[0];
    const int*   lat  = (const int*)d_in[1];
    const int*   lon  = (const int*)d_in[2];
    const float* elat = (const float*)d_in[3];
    const float* elon = (const float*)d_in[4];
    const float* pw   = (const float*)d_in[5];
    const float* pb   = (const float*)d_in[6];
    const float* Win  = (const float*)d_in[7];   // [4,2,1160,256]
    const float* cw   = (const float*)d_in[8];   // [4,2,640,4]
    const float* cb   = (const float*)d_in[9];   // [4,2,640]
    const float* dtb  = (const float*)d_in[10];  // [4,2,8]
    const float* alog = (const float*)d_in[11];  // [4,2,8]
    const float* Dp   = (const float*)d_in[12];  // [4,2,8]
    const float* nw   = (const float*)d_in[13];  // [4,2,512]
    const float* Wout = (const float*)d_in[14];  // [4,2,256,512]

    char* ws = (char*)d_ws;
    size_t off = 0;
    auto alloc = [&](size_t bytes) -> void* {
        void* p = ws + off;
        off = (off + bytes + 255) & ~(size_t)255;
        return p;
    };

    __bf16* Xbf    = (__bf16*)alloc((size_t)TOK * DMODEL * 2);
    __bf16* Winbf  = (__bf16*)alloc((size_t)NLAYERS * 2 * DPROJ * DMODEL * 2);
    __bf16* Woutbf = (__bf16*)alloc((size_t)NLAYERS * 2 * DMODEL * DINNER * 2);
    float*  zx     = (float*)alloc((size_t)TOK * DPROJ * 4);
    float*  xconv  = (float*)alloc((size_t)TOK * DXBC * 4);
    float*  dtbuf  = (float*)alloc((size_t)TOK * NHEADS * 4);
    float*  dAbuf  = (float*)alloc((size_t)TOK * NHEADS * 4);
    float*  ybuf   = (float*)alloc((size_t)TOK * DINNER * 4);
    __bf16* ybf    = (__bf16*)alloc((size_t)TOK * DINNER * 2);
    float*  xA     = (float*)alloc((size_t)TOK * DMODEL * 4);
    float*  xB     = (float*)alloc((size_t)TOK * DMODEL * 4);

    // one-time weight conversion (per call; deterministic)
    {
        int n1 = NLAYERS * 2 * DPROJ * DMODEL;
        cvt_bf16_kernel<<<(n1 + 255) / 256, 256, 0, stream>>>(Win, Winbf, n1);
        int n2 = NLAYERS * 2 * DMODEL * DINNER;
        cvt_bf16_kernel<<<(n2 + 255) / 256, 256, 0, stream>>>(Wout, Woutbf, n2);
    }

    // embedding / input projection
    embed_kernel<<<(TOK * DMODEL) / 256, 256, 0, stream>>>(feat, lat, lon, elat, elon, pw, pb, xA);

    float* xcur = xA;
    for (int layer = 0; layer < NLAYERS; ++layer) {
        float* xnext = (xcur == xA) ? xB : xA;
        cvt_bf16_kernel<<<(TOK * DMODEL) / 256, 256, 0, stream>>>(xcur, Xbf, TOK * DMODEL);

        for (int dir = 0; dir < 2; ++dir) {
            int idx = layer * 2 + dir;
            // in_proj: [TOK,256] x [1160,256]^T -> zx [TOK,1160]
            wmma_gemm_bf16<<<dim3(TOK / 128, (DPROJ + 63) / 64), 256, 0, stream>>>(
                Xbf, Winbf + (size_t)idx * DPROJ * DMODEL, zx, TOK, DPROJ, DMODEL, 0);
            // conv + silu
            conv_silu_kernel<<<(TOK * DXBC + 255) / 256, 256, 0, stream>>>(
                zx, cw + (size_t)idx * DXBC * DCONV, cb + (size_t)idx * DXBC, xconv, dir);
            // dt / dA
            dt_kernel<<<(TOK * NHEADS) / 256, 256, 0, stream>>>(
                zx, dtb + idx * NHEADS, alog + idx * NHEADS, dtbuf, dAbuf);
            // selective scan
            scan_kernel<<<BB * NHEADS, 256, 0, stream>>>(
                xconv, dtbuf, dAbuf, Dp + idx * NHEADS, ybuf, dir);
            // gate + RMSNorm -> bf16
            gate_norm_kernel<<<TOK, 256, 0, stream>>>(
                ybuf, zx, nw + (size_t)idx * DINNER, ybf);
            // out_proj: [TOK,512] x [256,512]^T -> xnext (fwd writes, bwd accumulates)
            wmma_gemm_bf16<<<dim3(TOK / 128, DMODEL / 64), 256, 0, stream>>>(
                ybf, Woutbf + (size_t)idx * DMODEL * DINNER, xnext, TOK, DMODEL, DINNER, dir);
        }
        xcur = xnext;
    }

    // outputs: x [4,4096,256] then mean over L [4,256]
    hipMemcpyAsync(d_out, xcur, (size_t)TOK * DMODEL * 4, hipMemcpyDeviceToDevice, stream);
    mean_kernel<<<BB * DMODEL, 256, 0, stream>>>(xcur, (float*)d_out + (size_t)TOK * DMODEL);
}
